// BatchHardTripletMarginLoss_51986284151331
// MI455X (gfx1250) — matrix operand order repaired and checked by reading, not compiled, not run on hardware
//
#include <hip/hip_runtime.h>
#include <math.h>

// ---------------- CDNA5 vector types ----------------
typedef __bf16 v8bf  __attribute__((ext_vector_type(8)));
typedef __bf16 v16bf __attribute__((ext_vector_type(16)));
typedef float  v8f   __attribute__((ext_vector_type(8)));

union V16 { v16bf v; v8bf h[2]; };

#define N_ROWS   4096
#define DIM      256
#define NWORDS   128            // 4096 bits / 32
#define INVALID  0x7fffffff
#define ROWPITCH 528            // 512B row + 16B pad -> lane bank stride 4, conflict-free
#define NSPLIT   8              // column splits (grid.y dimension folded into x)

// ---------------------------------------------------------------------------
// Kernel 1: row squared-norms + bf16 hi/lo split of embeddings.
// ---------------------------------------------------------------------------
__global__ void prep_kernel(const float* __restrict__ E,
                            float* __restrict__ sq,
                            unsigned short* __restrict__ ehi_u,
                            unsigned short* __restrict__ elo_u,
                            float* __restrict__ accum) {
    __bf16* ehi = (__bf16*)ehi_u;
    __bf16* elo = (__bf16*)elo_u;
    const int row = blockIdx.x;
    const int t   = threadIdx.x;            // 0..255 == DIM
    float v  = E[row * DIM + t];
    __bf16 h = (__bf16)v;
    float hf = (float)h;
    __bf16 l = (__bf16)(v - hf);
    ehi[row * DIM + t] = h;
    elo[row * DIM + t] = l;

    float s = v * v;
    #pragma unroll
    for (int off = 16; off; off >>= 1) s += __shfl_xor(s, off, 32);
    __shared__ float red[8];
    const int wid = t >> 5, ln = t & 31;
    if (ln == 0) red[wid] = s;
    __syncthreads();
    if (t == 0) {
        float tot = 0.f;
        #pragma unroll
        for (int i = 0; i < 8; ++i) tot += red[i];
        sq[row] = tot;
        if (row == 0) { accum[0] = 0.f; accum[1] = 0.f; }
    }
}

// ---------------------------------------------------------------------------
// Kernel 2: bit-pack boolean masks (word[m][w] bit b = mask[m][32w+b]).
// Masks are symmetric, so the mine kernel reads packed[n] for bits over m.
// ---------------------------------------------------------------------------
__global__ void pack_kernel(const unsigned char* __restrict__ pos,
                            const unsigned char* __restrict__ neg,
                            unsigned* __restrict__ ppos,
                            unsigned* __restrict__ pneg) {
    const int m   = blockIdx.x;
    const int wid = threadIdx.x >> 5;
    const int ln  = threadIdx.x & 31;
    const size_t rbase = (size_t)m * N_ROWS;
    for (int w = wid; w < NWORDS; w += 8) {
        unsigned pb = (pos[rbase + w * 32 + ln] != 0) ? (1u << ln) : 0u;
        unsigned nb = (neg[rbase + w * 32 + ln] != 0) ? (1u << ln) : 0u;
        #pragma unroll
        for (int off = 1; off < 32; off <<= 1) {
            pb |= __shfl_xor(pb, off, 32);
            nb |= __shfl_xor(nb, off, 32);
        }
        if (ln == 0) {
            ppos[m * NWORDS + w] = pb;
            pneg[m * NWORDS + w] = nb;
        }
    }
}

// ---------------------------------------------------------------------------
// Async stage of one 16-column B tile (hi+lo bf16 rows) into an LDS buffer.
// 16 KB -> 1024 x 16B chunks -> 4 GLOBAL_LOAD_ASYNC_TO_LDS_B128 per thread.
// Completion tracked with ASYNCcnt (s_wait_asynccnt + barrier by the caller).
// ---------------------------------------------------------------------------
__device__ __forceinline__ void stage_tile(unsigned char (*buf)[16 * ROWPITCH],
                                           const unsigned char* ehi_b,
                                           const unsigned char* elo_b,
                                           int ct, int tid) {
    #pragma unroll
    for (int j = 0; j < 4; ++j) {
        const int c  = tid + 256 * j;        // chunk id 0..1023
        const int p  = c >> 9;               // 0 = hi, 1 = lo
        const int r  = (c >> 5) & 15;        // row within tile
        const int kc = c & 31;               // 16B chunk within row
        const unsigned char* src =
            (p ? elo_b : ehi_b) + (size_t)(ct * 16 + r) * 512 + kc * 16;
        unsigned lds_off = (unsigned)(size_t)(void*)&buf[p][r * ROWPITCH + kc * 16];
        unsigned long long ga = (unsigned long long)(size_t)src;
        asm volatile("global_load_async_to_lds_b128 %0, %1, off"
                     :: "v"(lds_off), "v"(ga) : "memory");
    }
}

// ---------------------------------------------------------------------------
// Kernel 3: fused Gram-tile WMMA + batch-hard mining.
// grid = 32 row-blocks x 8 column-splits. Block = 8 waves; wave w owns rows
// [rb*128 + w*16, +16) across the block's 32 column tiles. B tiles staged
// once per block into double-buffered LDS via async loads; 24 WMMA per tile
// (3 error-compensated bf16 products x 8 K-steps), mining fused in epilogue.
// ---------------------------------------------------------------------------
__global__ void __launch_bounds__(256)
mine_kernel(const unsigned short* __restrict__ ehi_u,
            const unsigned short* __restrict__ elo_u,
            const float* __restrict__ sq,
            const unsigned* __restrict__ ppos,
            const unsigned* __restrict__ pneg,
            float* __restrict__ pPV, int* __restrict__ pPI,
            float* __restrict__ pNV, int* __restrict__ pNI) {
    const __bf16* ehi = (const __bf16*)ehi_u;
    const __bf16* elo = (const __bf16*)elo_u;
    const unsigned char* ehi_b = (const unsigned char*)ehi_u;
    const unsigned char* elo_b = (const unsigned char*)elo_u;

    const int rb  = blockIdx.x & 31;        // row block (128 rows)
    const int cs  = blockIdx.x >> 5;        // column split (32 tiles)
    const int tid = threadIdx.x;
    const int wid = tid >> 5;
    const int lane = tid & 31;
    const int l16  = lane & 15;
    const int hh   = lane >> 4;
    const int rowbase = rb * 128 + wid * 16;

    __shared__ __align__(16) unsigned char smem[2][2][16 * ROWPITCH]; // [buf][hi/lo]

    // --- preload A tiles (16x32 bf16: elems0..7 <- K=8h+0..7, elems8..15 <- K=16+8h+0..7)
    V16 Ahi[8], Alo[8];
    const int arow = rowbase + l16;
    #pragma unroll
    for (int ks = 0; ks < 8; ++ks) {
        const int k0 = ks * 32 + 8 * hh;
        Ahi[ks].h[0] = *(const v8bf*)(ehi + arow * DIM + k0);
        Ahi[ks].h[1] = *(const v8bf*)(ehi + arow * DIM + k0 + 16);
        Alo[ks].h[0] = *(const v8bf*)(elo + arow * DIM + k0);
        Alo[ks].h[1] = *(const v8bf*)(elo + arow * DIM + k0 + 16);
    }
    float sqm[8];
    #pragma unroll
    for (int v = 0; v < 8; ++v) sqm[v] = sq[rowbase + 8 * hh + v];

    float bestP[8], bestN[8];
    int   idxP[8], idxN[8];
    #pragma unroll
    for (int v = 0; v < 8; ++v) {
        bestP[v] = -INFINITY; bestN[v] = INFINITY;
        idxP[v] = INVALID;    idxN[v] = INVALID;
    }

    const int mword  = rowbase >> 5;
    const int mshift = (rowbase & 31) + 8 * hh;

    const int ct0 = cs * 32;
    // prefetch first tile into buffer 0
    stage_tile(smem[0], ehi_b, elo_b, ct0, tid);
    asm volatile("s_wait_asynccnt 0x0" ::: "memory");
    __syncthreads();

    for (int i = 0; i < 32; ++i) {
        const int ct  = ct0 + i;
        const int buf = i & 1;
        if (i + 1 < 32) stage_tile(smem[buf ^ 1], ehi_b, elo_b, ct + 1, tid);

        const int n = ct * 16 + l16;
        const float sqn = sq[n];
        const unsigned pw = ppos[n * NWORDS + mword] >> mshift;
        const unsigned nw = pneg[n * NWORDS + mword] >> mshift;

        const unsigned char* bhrow = &smem[buf][0][l16 * ROWPITCH];
        const unsigned char* blrow = &smem[buf][1][l16 * ROWPITCH];

        v8f chh = {0.f,0.f,0.f,0.f,0.f,0.f,0.f,0.f};
        v8f chl = {0.f,0.f,0.f,0.f,0.f,0.f,0.f,0.f};
        v8f clh = {0.f,0.f,0.f,0.f,0.f,0.f,0.f,0.f};
        #pragma unroll
        for (int ks = 0; ks < 8; ++ks) {
            const int k0b = (ks * 32 + 8 * hh) * 2;   // byte offset of K chunk
            V16 bh, bl;
            bh.h[0] = *(const v8bf*)(bhrow + k0b);
            bh.h[1] = *(const v8bf*)(bhrow + k0b + 32);
            bl.h[0] = *(const v8bf*)(blrow + k0b);
            bl.h[1] = *(const v8bf*)(blrow + k0b + 32);
            chh = __builtin_amdgcn_wmma_f32_16x16x32_bf16(false, Ahi[ks].v, false, bh.v,
                                                          (short)0, chh, false, false);
            chl = __builtin_amdgcn_wmma_f32_16x16x32_bf16(false, Ahi[ks].v, false, bl.v,
                                                          (short)0, chl, false, false);
            clh = __builtin_amdgcn_wmma_f32_16x16x32_bf16(false, Alo[ks].v, false, bh.v,
                                                          (short)0, clh, false, false);
        }
        #pragma unroll
        for (int v = 0; v < 8; ++v) {
            const float g   = chh[v] + chl[v] + clh[v];
            const float d2v = fmaxf(fmaf(-2.f, g, sqm[v] + sqn), 0.f);
            if ((pw >> v) & 1u) {
                if (d2v > bestP[v]) { bestP[v] = d2v; idxP[v] = n; }
            }
            if ((nw >> v) & 1u) {
                if (d2v < bestN[v]) { bestN[v] = d2v; idxN[v] = n; }
            }
        }
        asm volatile("s_wait_asynccnt 0x0" ::: "memory");
        __syncthreads();
    }

    // --- reduce across the 16 lanes of each half (same row set per half)
    #pragma unroll
    for (int v = 0; v < 8; ++v) {
        #pragma unroll
        for (int off = 1; off < 16; off <<= 1) {
            float ov = __shfl_xor(bestP[v], off, 32);
            int   oi = __shfl_xor(idxP[v],  off, 32);
            if (ov > bestP[v] || (ov == bestP[v] && oi < idxP[v])) { bestP[v] = ov; idxP[v] = oi; }
            float on = __shfl_xor(bestN[v], off, 32);
            int   oj = __shfl_xor(idxN[v],  off, 32);
            if (on < bestN[v] || (on == bestN[v] && oj < idxN[v])) { bestN[v] = on; idxN[v] = oj; }
        }
    }
    if (l16 == 0) {
        #pragma unroll
        for (int v = 0; v < 8; ++v) {
            const int row = rowbase + 8 * hh + v;
            const int o = cs * N_ROWS + row;
            pPV[o] = bestP[v]; pPI[o] = idxP[v];
            pNV[o] = bestN[v]; pNI[o] = idxN[v];
        }
    }
}

// ---------------------------------------------------------------------------
// Kernel 4: combine the 8 column-split partials per row (tie-break: min index).
// ---------------------------------------------------------------------------
__global__ void reduce_kernel(const float* __restrict__ pPV, const int* __restrict__ pPI,
                              const float* __restrict__ pNV, const int* __restrict__ pNI,
                              int* __restrict__ hpIdx, int* __restrict__ hnIdx) {
    const int row = blockIdx.x * blockDim.x + threadIdx.x;
    float bp = -INFINITY, bn = INFINITY;
    int ip = INVALID, in_ = INVALID;
    #pragma unroll
    for (int cs = 0; cs < NSPLIT; ++cs) {
        const int o = cs * N_ROWS + row;
        float pv = pPV[o]; int pi = pPI[o];
        if (pv > bp || (pv == bp && pi < ip)) { bp = pv; ip = pi; }
        float nv = pNV[o]; int ni = pNI[o];
        if (nv < bn || (nv == bn && ni < in_)) { bn = nv; in_ = ni; }
    }
    hpIdx[row] = ip;
    hnIdx[row] = in_;
}

// ---------------------------------------------------------------------------
// Kernel 5: exact fp32 distances for chosen (a, hp, hn) triplets + loss.
// ---------------------------------------------------------------------------
__global__ void finalize_kernel(const float* __restrict__ E,
                                const float* __restrict__ sq,
                                const int* __restrict__ hpIdx,
                                const int* __restrict__ hnIdx,
                                float* __restrict__ accum) {
    const int i    = blockIdx.x * (blockDim.x >> 5) + (threadIdx.x >> 5);
    const int lane = threadIdx.x & 31;
    if (i >= N_ROWS) return;
    const int hp = hpIdx[i], hn = hnIdx[i];
    if (hp == INVALID || hn == INVALID) return;

    const float* ei = E + (size_t)i  * DIM;
    const float* ep = E + (size_t)hp * DIM;
    const float* en = E + (size_t)hn * DIM;
    float ap = 0.f, an = 0.f, pn = 0.f;
    #pragma unroll
    for (int k = lane; k < DIM; k += 32) {
        const float a = ei[k], p = ep[k], g = en[k];
        ap = fmaf(a, p, ap);
        an = fmaf(a, g, an);
        pn = fmaf(p, g, pn);
    }
    #pragma unroll
    for (int off = 16; off; off >>= 1) {
        ap += __shfl_xor(ap, off, 32);
        an += __shfl_xor(an, off, 32);
        pn += __shfl_xor(pn, off, 32);
    }
    if (lane == 0) {
        const float dap = sqrtf(fmaxf(sq[i]  + sq[hp] - 2.f * ap, 0.f));
        const float dan = sqrtf(fmaxf(sq[i]  + sq[hn] - 2.f * an, 0.f));
        const float dpn = sqrtf(fmaxf(sq[hp] + sq[hn] - 2.f * pn, 0.f));
        const float l = dap - fminf(dan, dpn) + 0.2f;
        if (l > 0.f) {
            atomicAdd(&accum[0], l);
            atomicAdd(&accum[1], 1.f);
        }
    }
}

// Kernel 6: AvgNonZeroReducer scalar finish.
__global__ void finish_kernel(const float* __restrict__ accum, float* __restrict__ out) {
    if (threadIdx.x == 0 && blockIdx.x == 0) {
        const float cnt = accum[1];
        out[0] = (cnt > 0.f) ? (accum[0] / fmaxf(cnt, 1.f)) : 0.f;
    }
}

// ---------------------------------------------------------------------------
extern "C" void kernel_launch(void* const* d_in, const int* in_sizes, int n_in,
                              void* d_out, int out_size, void* d_ws, size_t ws_size,
                              hipStream_t stream) {
    (void)in_sizes; (void)n_in; (void)out_size; (void)ws_size;
    const float*         E   = (const float*)d_in[0];
    const unsigned char* pos = (const unsigned char*)d_in[1];
    const unsigned char* neg = (const unsigned char*)d_in[2];
    float* out = (float*)d_out;

    char* ws = (char*)d_ws;
    // workspace layout (bytes)
    size_t off = 0;
    float*          sq    = (float*)(ws + off);          off += 16384;
    unsigned short* ehi   = (unsigned short*)(ws + off); off += 2097152;
    unsigned short* elo   = (unsigned short*)(ws + off); off += 2097152;
    unsigned*       ppos  = (unsigned*)(ws + off);       off += 2097152;
    unsigned*       pneg  = (unsigned*)(ws + off);       off += 2097152;
    float*          pPV   = (float*)(ws + off);          off += NSPLIT * N_ROWS * 4;
    int*            pPI   = (int*)(ws + off);            off += NSPLIT * N_ROWS * 4;
    float*          pNV   = (float*)(ws + off);          off += NSPLIT * N_ROWS * 4;
    int*            pNI   = (int*)(ws + off);            off += NSPLIT * N_ROWS * 4;
    int*            hpI   = (int*)(ws + off);            off += 16384;
    int*            hnI   = (int*)(ws + off);            off += 16384;
    float*          accum = (float*)(ws + off);          off += 64;

    prep_kernel<<<N_ROWS, 256, 0, stream>>>(E, sq, ehi, elo, accum);
    pack_kernel<<<N_ROWS, 256, 0, stream>>>(pos, neg, ppos, pneg);
    mine_kernel<<<32 * NSPLIT, 256, 0, stream>>>(ehi, elo, sq, ppos, pneg,
                                                 pPV, pPI, pNV, pNI);
    reduce_kernel<<<N_ROWS / 256, 256, 0, stream>>>(pPV, pPI, pNV, pNI, hpI, hnI);
    finalize_kernel<<<N_ROWS / 8, 256, 0, stream>>>(E, sq, hpI, hnI, accum);
    finish_kernel<<<1, 64, 0, stream>>>(accum, out);
}
